// CellSegmentationLoss_28810640621718
// MI455X (gfx1250) — compile-verified, matrix operand order checked by reading, not run on hardware
//
#include <hip/hip_runtime.h>

typedef __attribute__((ext_vector_type(16))) _Float16 v16h;
typedef __attribute__((ext_vector_type(8)))  float    v8f;

// Problem geometry (fixed by the reference)
constexpr int  NIMG          = 16;
constexpr int  PIX_PER_IMG   = 1024 * 1024;
constexpr int  BPI           = 128;                       // blocks per image
constexpr int  THREADS       = 256;
constexpr int  PIX_PER_BLOCK = PIX_PER_IMG / BPI;         // 8192
constexpr int  VEC_ITERS     = PIX_PER_BLOCK / (THREADS * 4); // 8 float4 iters/thread
constexpr int  WS_FLOATS     = 2 + NIMG * 5;              // 82

// ws layout: [0]=sum_focal, [1]=sum_ce, then per image b: {pt, p, t, bin*t, bin}

__global__ void CellSegLoss_zero_ws(float* __restrict__ ws) {
    int i = threadIdx.x;
    if (i < WS_FLOATS) ws[i] = 0.0f;
}

__global__ __launch_bounds__(THREADS)
void CellSegLoss_main(const float* __restrict__ pred,
                      const float* __restrict__ gt,
                      float* __restrict__ ws) {
    const int img   = blockIdx.x / BPI;
    const int chunk = blockIdx.x % BPI;
    const long base = (long)img * PIX_PER_IMG + (long)chunk * PIX_PER_BLOCK;
    const float4* __restrict__ p4 = (const float4*)(pred + base);
    const float4* __restrict__ t4 = (const float4*)(gt + base);
    const int tid = threadIdx.x;

    float a_f = 0.f, a_ce = 0.f, a_pt = 0.f, a_p = 0.f,
          a_t = 0.f, a_bt = 0.f, a_b = 0.f;

    #pragma unroll
    for (int it = 0; it < VEC_ITERS; ++it) {
        float4 xv = p4[it * THREADS + tid];   // coalesced global_load_b128
        float4 tv = t4[it * THREADS + tid];
        #pragma unroll
        for (int c = 0; c < 4; ++c) {
            float x = ((const float*)&xv)[c];
            float t = ((const float*)&tv)[c];
            float ax = fabsf(x);
            float e  = __expf(-ax);                      // v_exp_f32
            float sp = fmaxf(x, 0.f) + __logf(1.0f + e); // stable softplus, v_log_f32
            float ce = sp - x * t;                       // BCE-with-logits
            float r  = 1.0f / (1.0f + e);                // single reciprocal
            float p  = (x >= 0.f) ? r : e * r;           // sigmoid (v_cndmask)
            float pt = p * t;
            float p_t     = pt + (1.f - p) * (1.f - t);
            float one_m   = 1.f - p_t;
            float alpha_t = 0.75f - 0.5f * t;            // 0.25*t + 0.75*(1-t)
            a_f  += alpha_t * ce * one_m * one_m;        // focal term
            a_ce += ce;
            a_pt += pt;
            a_p  += p;
            a_t  += t;
            float bin = (p > 0.5f) ? 1.f : 0.f;
            a_b  += bin;
            a_bt += bin * t;
        }
    }

    // Stage partials: quantity-major [8][256] (row 7 zeroed)
    __shared__ float lds[8 * THREADS];
    lds[0 * THREADS + tid] = a_f;
    lds[1 * THREADS + tid] = a_ce;
    lds[2 * THREADS + tid] = a_pt;
    lds[3 * THREADS + tid] = a_p;
    lds[4 * THREADS + tid] = a_t;
    lds[5 * THREADS + tid] = a_bt;
    lds[6 * THREADS + tid] = a_b;
    lds[7 * THREADS + tid] = 0.0f;
    __syncthreads();

    // Wave 0: reduce 256 partials x 7 quantities with 8 chained WMMAs.
    // A = all-ones (16x32 f16); B column n carries quantity (n & 7) partials
    // along K (thread index) -- columns 8..15 are redundant duplicates, which
    // keeps every lane's LDS reads unconditional and vectorizable.
    // fp32 C accumulation is exact; since A is all ones the result is
    // invariant to the exact K permutation of the B layout.
    if (tid < 32) {
        v16h A;
        #pragma unroll
        for (int j = 0; j < 16; ++j) A[j] = (_Float16)1.0f;
        v8f C = {};

        const int q = tid & 7;
        const float* src = lds + q * THREADS + ((tid >> 4) << 4); // 64B aligned
        #pragma unroll
        for (int i = 0; i < 8; ++i) {
            const float4* s4 = (const float4*)(src + i * 32);     // ds_load_b128 x4
            float4 f0 = s4[0], f1 = s4[1], f2 = s4[2], f3 = s4[3];
            v16h Bv;
            Bv[0]  = (_Float16)f0.x; Bv[1]  = (_Float16)f0.y;
            Bv[2]  = (_Float16)f0.z; Bv[3]  = (_Float16)f0.w;
            Bv[4]  = (_Float16)f1.x; Bv[5]  = (_Float16)f1.y;
            Bv[6]  = (_Float16)f1.z; Bv[7]  = (_Float16)f1.w;
            Bv[8]  = (_Float16)f2.x; Bv[9]  = (_Float16)f2.y;
            Bv[10] = (_Float16)f2.z; Bv[11] = (_Float16)f2.w;
            Bv[12] = (_Float16)f3.x; Bv[13] = (_Float16)f3.y;
            Bv[14] = (_Float16)f3.z; Bv[15] = (_Float16)f3.w;
            C = __builtin_amdgcn_wmma_f32_16x16x32_f16(
                    /*neg_a=*/false, A, /*neg_b=*/false, Bv,
                    /*c_mod=*/(short)0, C, /*reuse_a=*/false, /*reuse_b=*/false);
        }
        // D row 0: lane n holds the total for quantity (n & 7)
        float s = C[0];
        if (tid < 7) {
            float* dst;
            if (tid == 0)      dst = ws + 0;                       // focal
            else if (tid == 1) dst = ws + 1;                       // ce
            else               dst = ws + 2 + img * 5 + (tid - 2); // pt,p,t,bint,bin
            atomicAdd(dst, s);
        }
    }
}

__global__ void CellSegLoss_finalize(const float* __restrict__ ws,
                                     const float* __restrict__ pred_iou,
                                     float* __restrict__ out) {
    if (threadIdx.x != 0 || blockIdx.x != 0) return;
    const float S = 1e-6f;
    const float N = 16.0f * 1024.0f * 1024.0f;     // total pixels
    float focal   = ws[0] / N;
    float ce_mean = ws[1] / N;
    float dice_acc = 0.f, iou_acc = 0.f;
    for (int b = 0; b < NIMG; ++b) {
        const float* w = ws + 2 + b * 5;
        float spt = w[0], sp = w[1], st = w[2], sbt = w[3], sb = w[4];
        dice_acc += (2.0f * spt + S) / (sp + st + S);
        float inter  = sbt;
        float uni    = sb + st - sbt;
        float actual = (inter + S) / (uni + S);
        float d = pred_iou[b] - actual;
        iou_acc += d * d;
    }
    float dice     = 1.0f - dice_acc / (float)NIMG;
    float boundary = 2.0f * ce_mean;               // uniform weight map (see ref)
    float total    = focal + dice + 0.5f * boundary;
    out[0] = total + 0.1f * (iou_acc / (float)NIMG);
}

extern "C" void kernel_launch(void* const* d_in, const int* in_sizes, int n_in,
                              void* d_out, int out_size, void* d_ws, size_t ws_size,
                              hipStream_t stream) {
    (void)in_sizes; (void)n_in; (void)out_size; (void)ws_size;
    const float* pred = (const float*)d_in[0];   // (16,1,1024,1024) fp32 logits
    const float* gt   = (const float*)d_in[1];   // (16,1,1024,1024) fp32 {0,1}
    const float* piou = (const float*)d_in[2];   // (16,1) fp32
    float* ws  = (float*)d_ws;
    float* out = (float*)d_out;

    CellSegLoss_zero_ws<<<1, 128, 0, stream>>>(ws);
    CellSegLoss_main<<<NIMG * BPI, THREADS, 0, stream>>>(pred, gt, ws);
    CellSegLoss_finalize<<<1, 32, 0, stream>>>(ws, piou, out);
}
